// SpatialTransformer_47519518163203
// MI455X (gfx1250) — matrix operand order repaired and checked by reading, not compiled, not run on hardware
//
#include <hip/hip_runtime.h>

typedef unsigned short u16;
typedef __attribute__((ext_vector_type(16))) __bf16 v16bf;
typedef __attribute__((ext_vector_type(8)))  float  v8f;
typedef __attribute__((ext_vector_type(4)))  unsigned int v4u;
typedef __attribute__((ext_vector_type(8)))  int v8i;
typedef __attribute__((ext_vector_type(4)))  int v4i;

union FragU { uint4 u[2]; v16bf v; };

__device__ __forceinline__ u16 f2bf(float f) {
    unsigned u = __float_as_uint(f);
    unsigned r = (u + 0x7fffu + ((u >> 16) & 1u)) >> 16;
    return (u16)r;
}

__device__ __forceinline__ v8f wmma_bf16(v16bf a, v16bf b, v8f c) {
    return __builtin_amdgcn_wmma_f32_16x16x32_bf16(false, a, false, b, (short)0, c, false, false);
}

// A-operand fragment: 16x32 (MxK). lane<16: M=lane, K in {0..7,16..23};
// lane>=16: M=lane-16, K in {8..15,24..31}. tile row-major (rows x stride halfs).
__device__ __forceinline__ v16bf ld_frag_a(const u16* tile, int row0, int stride) {
    int lane = threadIdx.x & 31;
    const u16* p = tile + (row0 + (lane & 15)) * stride + ((lane >> 4) << 3);
    FragU f;
    f.u[0] = *(const uint4*)p;
    f.u[1] = *(const uint4*)(p + 16);
    return f.v;
}

// B-operand fragment: 32x16 (KxN) from an (N x K) row-major tile.
// lane<16: N=lane, K in {0..7,16..23}; lane>=16: N=lane-16, K in {8..15,24..31}.
__device__ __forceinline__ v16bf ld_frag_b(const u16* tile, int row0, int stride) {
    int lane = threadIdx.x & 31;
    const u16* p = tile + (row0 + (lane & 15)) * stride + ((lane >> 4) << 3);
    FragU f;
    f.u[0] = *(const uint4*)p;
    f.u[1] = *(const uint4*)(p + 16);
    return f.v;
}

// LDS byte offset of a __shared__ pointer: flat addr low 32 bits (ISA 10.2 LDS aperture).
__device__ __forceinline__ unsigned lds_off(const void* p) {
    return (unsigned)(unsigned long long)(size_t)p;
}

// TDM 2D tile load: tile_h rows x tile_w bf16 elements, global row stride = stride
// elements, into contiguous row-major LDS at lds_byte_off. D# packed per ISA 8.3/8.4:
// group0 = {count=1, lds_addr, global_addr[56:0], type=2}; group1 = {mask=0,
// data_size=1(2B), tensor_dim0/1 = tile dims (no OOB clip), tile_dim0/1, stride0}.
__device__ __forceinline__ void tdm_load_2d(unsigned lds_byte_off, const void* gptr,
                                            unsigned tile_w, unsigned tile_h,
                                            unsigned stride) {
    unsigned long long ga = (unsigned long long)(size_t)gptr;
    v4u g0;
    g0[0] = 1u;                                              // count=1, user D#
    g0[1] = lds_byte_off;                                    // lds_addr (bytes)
    g0[2] = (unsigned)ga;                                    // global_addr[31:0]
    g0[3] = (unsigned)((ga >> 32) & 0x01FFFFFFu) | 0x80000000u; // [56:32] | type=2
    v8i g1;
    g1[0] = (int)(1u << 16);                                 // data_size=1 -> 2 bytes
    g1[1] = (int)((tile_w & 0xFFFFu) << 16);                 // tensor_dim0[15:0]
    g1[2] = (int)((tile_w >> 16) | ((tile_h & 0xFFFFu) << 16)); // td0 hi | td1 lo
    g1[3] = (int)((tile_h >> 16) | (tile_w << 16));          // td1 hi | tile_dim0
    g1[4] = (int)tile_h;                                     // tile_dim1 (tile_dim2=0)
    g1[5] = (int)stride;                                     // tensor_dim0_stride[31:0]
    g1[6] = 0;
    g1[7] = 0;
    v4i z4 = {0, 0, 0, 0};
#if defined(__clang_major__) && (__clang_major__ >= 23)
    v8i z8 = {0, 0, 0, 0, 0, 0, 0, 0};
    __builtin_amdgcn_tensor_load_to_lds(g0, g1, z4, z4, z8, 0);
#else
    __builtin_amdgcn_tensor_load_to_lds(g0, g1, z4, z4, 0);
#endif
}

__device__ __forceinline__ void block_sum2(float& s, float& s2, float* sh) {
    #pragma unroll
    for (int m = 1; m < 32; m <<= 1) {
        s  += __shfl_xor(s,  m, 32);
        s2 += __shfl_xor(s2, m, 32);
    }
    int w = threadIdx.x >> 5, lane = threadIdx.x & 31;
    if (lane == 0) { sh[w] = s; sh[8 + w] = s2; }
    __syncthreads();
    if (threadIdx.x < 32) {
        float a = (lane < 8) ? sh[lane] : 0.f;
        float b = (lane < 8) ? sh[8 + lane] : 0.f;
        #pragma unroll
        for (int m = 1; m < 8; m <<= 1) {
            a += __shfl_xor(a, m, 32);
            b += __shfl_xor(b, m, 32);
        }
        if (lane == 0) { sh[0] = a; sh[8] = b; }
    }
    __syncthreads();
    s = sh[0];
    s2 = sh[8];
}

// ---------------- weight convert (optional transpose) f32 -> bf16 ----------------
__global__ void st_convert_w(const float* __restrict__ src, u16* __restrict__ dst,
                             int N, int K, int transpose) {
    int i = blockIdx.x * 256 + threadIdx.x;
    if (i >= N * K) return;
    int n = i / K, k = i - n * K;
    float v = transpose ? src[(size_t)k * N + n] : src[i];
    dst[i] = f2bf(v);
}

// ---------------- GroupNorm: x (B,C,H,W) f32 -> (B,S,C) bf16 transposed ----------
__global__ void st_groupnorm(const float* __restrict__ x, const float* __restrict__ gw,
                             const float* __restrict__ gb, u16* __restrict__ out) {
    __shared__ float sh[16];
    int bg = blockIdx.x;
    int b = bg >> 3, g = bg & 7;
    const float* base = x + ((size_t)(b * 512 + g * 64) << 10);
    int tid = threadIdx.x;
    float s = 0.f, s2 = 0.f;
    for (int i = tid; i < 65536; i += 256) {
        float v = base[i];
        s += v; s2 += v * v;
    }
    block_sum2(s, s2, sh);
    float mu = s * (1.f / 65536.f);
    float var = s2 * (1.f / 65536.f) - mu * mu;
    float rs = rsqrtf(var + 1e-5f);
    for (int i = tid; i < 65536; i += 256) {
        int cl = i >> 10, sp = i & 1023;
        int c = g * 64 + cl;
        float v = (base[i] - mu) * rs * gw[c] + gb[c];
        out[((size_t)(b * 1024 + sp)) * 512 + c] = f2bf(v);
    }
}

// ---------------- LayerNorm: t (M,512) f32 -> tn (M,512) bf16 --------------------
__global__ void st_layernorm(const float* __restrict__ x, const float* __restrict__ w,
                             const float* __restrict__ b, u16* __restrict__ out) {
    __shared__ float sh[16];
    size_t row = blockIdx.x;
    int tid = threadIdx.x;
    const float* xr = x + row * 512;
    float v0 = xr[tid], v1 = xr[tid + 256];
    float s = v0 + v1, s2 = v0 * v0 + v1 * v1;
    block_sum2(s, s2, sh);
    float mu = s * (1.f / 512.f);
    float var = s2 * (1.f / 512.f) - mu * mu;
    float rs = rsqrtf(var + 1e-5f);
    out[row * 512 + tid]       = f2bf((v0 - mu) * rs * w[tid] + b[tid]);
    out[row * 512 + tid + 256] = f2bf((v1 - mu) * rs * w[tid + 256] + b[tid + 256]);
}

// ---------------- generic bf16 GEMM: D[m,n] = sum_k A[m,k]*W[n,k] ----------------
// 128x128 block tile, 8 waves (4x2), wave tile 32x64, K-step 32.
// TDM double-buffered staging: wave 0 issues tensor_load_to_lds for tile i+1,
// s_wait_tensorcnt guarantees tile i resident (TENSORcnt in-order per wave).
// epi: 0 bf16 store; 1 f32 +bias; 2 f32 +bias+residual(in-place); 3 = 2 + bf16 copy;
//      4 pout transposed store + bias + x residual.
__global__ void st_gemm(const u16* __restrict__ A, const u16* __restrict__ W,
                        const float* __restrict__ bias, float* __restrict__ outF,
                        u16* __restrict__ outH, const float* __restrict__ resx,
                        int K, int Nt, int epi) {
    __shared__ u16 sA[2][128 * 32];
    __shared__ u16 sB[2][128 * 32];
    int tid = threadIdx.x, lane = tid & 31, wid = tid >> 5;
    int wm = wid >> 1, wn = wid & 1;
    int mt = blockIdx.x * 128;
    int nt = blockIdx.y * 128;
    int nk = K >> 5;

    v8f acc[2][4];
    #pragma unroll
    for (int i = 0; i < 2; ++i)
        #pragma unroll
        for (int j = 0; j < 4; ++j) acc[i][j] = (v8f)0.f;

    const u16* gA = A + (size_t)mt * K;
    const u16* gB = W + (size_t)nt * K;
    if (wid == 0) {
        tdm_load_2d(lds_off(sA[0]), gA, 32, 128, K);
        tdm_load_2d(lds_off(sB[0]), gB, 32, 128, K);
    }
    for (int it = 0; it < nk; ++it) {
        if (wid == 0) {
            if (it + 1 < nk) {
                int k0n = (it + 1) << 5;
                tdm_load_2d(lds_off(sA[(it + 1) & 1]), gA + k0n, 32, 128, K);
                tdm_load_2d(lds_off(sB[(it + 1) & 1]), gB + k0n, 32, 128, K);
                __builtin_amdgcn_s_wait_tensorcnt(2);   // tile `it` complete
            } else {
                __builtin_amdgcn_s_wait_tensorcnt(0);
            }
        }
        __syncthreads();
        const u16* cA = sA[it & 1];
        const u16* cB = sB[it & 1];
        v16bf a0 = ld_frag_a(cA, wm * 32, 32);
        v16bf a1 = ld_frag_a(cA, wm * 32 + 16, 32);
        #pragma unroll
        for (int j = 0; j < 4; ++j) {
            v16bf b = ld_frag_b(cB, wn * 64 + j * 16, 32);
            acc[0][j] = wmma_bf16(a0, b, acc[0][j]);
            acc[1][j] = wmma_bf16(a1, b, acc[1][j]);
        }
        __syncthreads();   // all waves done with tile `it` before TDM overwrites it
    }

    int colb = lane & 15;
    int rowb = (lane >> 4) << 3;
    #pragma unroll
    for (int i = 0; i < 2; ++i) {
        #pragma unroll
        for (int j = 0; j < 4; ++j) {
            int col = nt + wn * 64 + j * 16 + colb;
            float bv = (epi != 0) ? bias[col] : 0.f;
            #pragma unroll
            for (int e = 0; e < 8; ++e) {
                int row = mt + wm * 32 + i * 16 + rowb + e;
                float vv = acc[i][j][e];
                if (epi == 0) {
                    outH[(size_t)row * Nt + col] = f2bf(vv);
                } else if (epi == 1) {
                    outF[(size_t)row * Nt + col] = vv + bv;
                } else if (epi == 2) {
                    size_t ix = (size_t)row * Nt + col;
                    outF[ix] = vv + bv + outF[ix];
                } else if (epi == 3) {
                    size_t ix = (size_t)row * Nt + col;
                    float r = vv + bv + outF[ix];
                    outF[ix] = r;
                    outH[ix] = f2bf(r);
                } else {
                    int bb = row >> 10, ss = row & 1023;
                    size_t ix = (((size_t)(bb * 512 + col)) << 10) + ss;
                    outF[ix] = vv + bv + resx[ix];
                }
            }
        }
    }
}

// ---------------- fused gated MLP GEMM: m = (tn@Wa+ba) * gelu(tn@Wg+bg) ----------
// 128(M) x 64(N of m) block tile, waves 4x2, wave tile 32x32, two accumulator sets.
// TDM double-buffered staging of A / Wa / Wg tiles.
__global__ void st_gemm_gated(const u16* __restrict__ A, const u16* __restrict__ Wt,
                              const float* __restrict__ bgv, u16* __restrict__ outM) {
    const int K = 512;
    __shared__ u16 sA[2][128 * 32];
    __shared__ u16 sBa[2][64 * 32];
    __shared__ u16 sBg[2][64 * 32];
    int tid = threadIdx.x, lane = tid & 31, wid = tid >> 5;
    int wm = wid >> 1, wn = wid & 1;
    int mt = blockIdx.x * 128;
    int nt = blockIdx.y * 64;
    const int nk = K >> 5;

    v8f aA[2][2], aG[2][2];
    #pragma unroll
    for (int i = 0; i < 2; ++i)
        #pragma unroll
        for (int j = 0; j < 2; ++j) { aA[i][j] = (v8f)0.f; aG[i][j] = (v8f)0.f; }

    const u16* gA  = A + (size_t)mt * K;
    const u16* gBa = Wt + (size_t)nt * K;
    const u16* gBg = Wt + (size_t)(2048 + nt) * K;
    if (wid == 0) {
        tdm_load_2d(lds_off(sA[0]),  gA,  32, 128, K);
        tdm_load_2d(lds_off(sBa[0]), gBa, 32, 64,  K);
        tdm_load_2d(lds_off(sBg[0]), gBg, 32, 64,  K);
    }
    for (int it = 0; it < nk; ++it) {
        if (wid == 0) {
            if (it + 1 < nk) {
                int k0n = (it + 1) << 5;
                tdm_load_2d(lds_off(sA[(it + 1) & 1]),  gA + k0n,  32, 128, K);
                tdm_load_2d(lds_off(sBa[(it + 1) & 1]), gBa + k0n, 32, 64,  K);
                tdm_load_2d(lds_off(sBg[(it + 1) & 1]), gBg + k0n, 32, 64,  K);
                __builtin_amdgcn_s_wait_tensorcnt(3);
            } else {
                __builtin_amdgcn_s_wait_tensorcnt(0);
            }
        }
        __syncthreads();
        const u16* cA  = sA[it & 1];
        const u16* cBa = sBa[it & 1];
        const u16* cBg = sBg[it & 1];
        v16bf a0 = ld_frag_a(cA, wm * 32, 32);
        v16bf a1 = ld_frag_a(cA, wm * 32 + 16, 32);
        #pragma unroll
        for (int j = 0; j < 2; ++j) {
            v16bf ba = ld_frag_b(cBa, wn * 32 + j * 16, 32);
            v16bf bg = ld_frag_b(cBg, wn * 32 + j * 16, 32);
            aA[0][j] = wmma_bf16(a0, ba, aA[0][j]);
            aA[1][j] = wmma_bf16(a1, ba, aA[1][j]);
            aG[0][j] = wmma_bf16(a0, bg, aG[0][j]);
            aG[1][j] = wmma_bf16(a1, bg, aG[1][j]);
        }
        __syncthreads();
    }

    int colb = lane & 15;
    int rowb = (lane >> 4) << 3;
    #pragma unroll
    for (int i = 0; i < 2; ++i) {
        #pragma unroll
        for (int j = 0; j < 2; ++j) {
            int col = nt + wn * 32 + j * 16 + colb;
            float ba_ = bgv[col], bg_ = bgv[2048 + col];
            #pragma unroll
            for (int e = 0; e < 8; ++e) {
                int row = mt + wm * 32 + i * 16 + rowb + e;
                float a = aA[i][j][e] + ba_;
                float g = aG[i][j][e] + bg_;
                float gl = 0.5f * g * (1.f + erff(g * 0.70710678118654752f));
                outM[(size_t)row * 2048 + col] = f2bf(a * gl);
            }
        }
    }
}

// ---------------- flash attention: q,k,v (B,S,C) bf16 -> out (B,S,C) bf16 --------
// grid (S/128, NH, B), 256 threads, each wave owns 16 query rows, KV-blocks of 32.
__global__ void st_attn(const u16* __restrict__ q, const u16* __restrict__ k,
                        const u16* __restrict__ v, u16* __restrict__ out) {
    const int S = 1024, C = 512;
    __shared__ u16 sK[32 * 64];
    __shared__ u16 sVt[64 * 32];
    __shared__ u16 sP[8 * 16 * 32];
    int qt = blockIdx.x, h = blockIdx.y, b = blockIdx.z;
    int tid = threadIdx.x, lane = tid & 31, wid = tid >> 5;
    u16* wP = sP + wid * (16 * 32);
    size_t rbase = (size_t)b * S;
    int hoff = h * 64;
    int q0 = qt * 128 + wid * 16;

    v16bf qa0, qa1;
    {
        const u16* qp = q + (rbase + q0 + (lane & 15)) * C + hoff + ((lane >> 4) << 3);
        FragU f;
        f.u[0] = *(const uint4*)qp;        f.u[1] = *(const uint4*)(qp + 16);
        qa0 = f.v;
        f.u[0] = *(const uint4*)(qp + 32); f.u[1] = *(const uint4*)(qp + 48);
        qa1 = f.v;
    }

    v8f acc[4];
    #pragma unroll
    for (int j = 0; j < 4; ++j) acc[j] = (v8f)0.f;
    float rm[8], rs_[8];
    #pragma unroll
    for (int e = 0; e < 8; ++e) { rm[e] = -3.0e38f; rs_[e] = 0.f; }

    int r = tid >> 3, c8 = (tid & 7) << 3;
    for (int kb = 0; kb < S; kb += 32) {
        // stage K block and V^T block; both globals in flight before LDS stores
        uint4 kreg = *(const uint4*)(k + (rbase + kb + r) * C + hoff + c8);
        union { uint4 u; u16 h[8]; } tv;
        tv.u = *(const uint4*)(v + (rbase + kb + r) * C + hoff + c8);
        *(uint4*)&sK[r * 64 + c8] = kreg;
        #pragma unroll
        for (int ii = 0; ii < 8; ++ii) sVt[(c8 + ii) * 32 + r] = tv.h[ii];
        __syncthreads();

        v8f d0v = (v8f)0.f, d1v = (v8f)0.f;
        d0v = wmma_bf16(qa0, ld_frag_b(sK,       0, 64), d0v);
        d0v = wmma_bf16(qa1, ld_frag_b(sK + 32,  0, 64), d0v);
        d1v = wmma_bf16(qa0, ld_frag_b(sK,      16, 64), d1v);
        d1v = wmma_bf16(qa1, ld_frag_b(sK + 32, 16, 64), d1v);

        #pragma unroll
        for (int e = 0; e < 8; ++e) {
            float s0 = d0v[e] * 0.125f, s1 = d1v[e] * 0.125f;
            float mx = fmaxf(s0, s1);
            mx = fmaxf(mx, __shfl_xor(mx, 1, 32));
            mx = fmaxf(mx, __shfl_xor(mx, 2, 32));
            mx = fmaxf(mx, __shfl_xor(mx, 4, 32));
            mx = fmaxf(mx, __shfl_xor(mx, 8, 32));
            float nm = fmaxf(rm[e], mx);
            float al = expf(rm[e] - nm);
            float p0 = expf(s0 - nm), p1 = expf(s1 - nm);
            float ps = p0 + p1;
            ps += __shfl_xor(ps, 1, 32);
            ps += __shfl_xor(ps, 2, 32);
            ps += __shfl_xor(ps, 4, 32);
            ps += __shfl_xor(ps, 8, 32);
            rs_[e] = rs_[e] * al + ps;
            rm[e] = nm;
            acc[0][e] *= al; acc[1][e] *= al; acc[2][e] *= al; acc[3][e] *= al;
            int prow = ((lane >> 4) << 3) + e;
            wP[prow * 32 + (lane & 15)]      = f2bf(p0);
            wP[prow * 32 + 16 + (lane & 15)] = f2bf(p1);
        }

        v16bf pa = ld_frag_a(wP, 0, 32);
        #pragma unroll
        for (int dt = 0; dt < 4; ++dt) {
            v16bf bv = ld_frag_b(sVt, dt * 16, 32);
            acc[dt] = wmma_bf16(pa, bv, acc[dt]);
        }
        __syncthreads();
    }

    #pragma unroll
    for (int dt = 0; dt < 4; ++dt) {
        #pragma unroll
        for (int e = 0; e < 8; ++e) {
            int row = q0 + ((lane >> 4) << 3) + e;
            float val = acc[dt][e] / rs_[e];
            out[(rbase + row) * C + hoff + dt * 16 + (lane & 15)] = f2bf(val);
        }
    }
}

// ------------------------------- launch --------------------------------------
extern "C" void kernel_launch(void* const* d_in, const int* in_sizes, int n_in,
                              void* d_out, int out_size, void* d_ws, size_t ws_size,
                              hipStream_t stream) {
    const float* x      = (const float*)d_in[0];
    const float* gn_w   = (const float*)d_in[1];
    const float* gn_b   = (const float*)d_in[2];
    const float* pin_w  = (const float*)d_in[3];
    const float* pin_b  = (const float*)d_in[4];
    const float* ln1_w  = (const float*)d_in[5];
    const float* ln1_b  = (const float*)d_in[6];
    const float* wq     = (const float*)d_in[7];
    const float* wk     = (const float*)d_in[8];
    const float* wv_in  = (const float*)d_in[9];
    const float* wo     = (const float*)d_in[10];
    const float* bo     = (const float*)d_in[11];
    const float* ln3_w  = (const float*)d_in[12];
    const float* ln3_b  = (const float*)d_in[13];
    const float* wg     = (const float*)d_in[14];
    const float* bg     = (const float*)d_in[15];
    const float* wd     = (const float*)d_in[16];
    const float* bd     = (const float*)d_in[17];
    const float* pout_w = (const float*)d_in[18];
    const float* pout_b = (const float*)d_in[19];

    char* ws = (char*)d_ws;
    u16* pinW  = (u16*)ws;
    u16* wqT   = pinW + 262144;
    u16* wkT   = pinW + 524288;
    u16* wvT   = pinW + 786432;
    u16* woT   = pinW + 1048576;
    u16* wgT   = pinW + 1310720;   // 4096 x 512
    u16* wdT   = pinW + 3407872;   // 512 x 2048
    u16* poutW = pinW + 4456448;   // end 4718592 halfs = 9437184 B
    const size_t OFF_T = 9437184;
    float* tbuf = (float*)(ws + OFF_T);                         // 16384x512 f32
    u16*  abuf  = (u16*)(ws + OFF_T + 33554432);                // 16384x512 bf16
    u16*  qbuf  = (u16*)(ws + OFF_T + 33554432 + 16777216);
    u16*  kbuf  = (u16*)(ws + OFF_T + 33554432 + 2 * 16777216);
    u16*  vbuf  = (u16*)(ws + OFF_T + 33554432 + 3 * 16777216);
    u16*  mbuf  = (u16*)(ws + OFF_T + 33554432 + 4 * 16777216); // 16384x2048 bf16

    st_convert_w<<<1024, 256, 0, stream>>>(pin_w,  pinW,  512,  512,  0);
    st_convert_w<<<1024, 256, 0, stream>>>(wq,     wqT,   512,  512,  1);
    st_convert_w<<<1024, 256, 0, stream>>>(wk,     wkT,   512,  512,  1);
    st_convert_w<<<1024, 256, 0, stream>>>(wv_in,  wvT,   512,  512,  1);
    st_convert_w<<<1024, 256, 0, stream>>>(wo,     woT,   512,  512,  1);
    st_convert_w<<<8192, 256, 0, stream>>>(wg,     wgT,   4096, 512,  1);
    st_convert_w<<<4096, 256, 0, stream>>>(wd,     wdT,   512,  2048, 1);
    st_convert_w<<<1024, 256, 0, stream>>>(pout_w, poutW, 512,  512,  0);

    st_groupnorm<<<128, 256, 0, stream>>>(x, gn_w, gn_b, abuf);

    st_gemm<<<dim3(128, 4), 256, 0, stream>>>(abuf, pinW, pin_b, tbuf, nullptr, nullptr, 512, 512, 1);

    st_layernorm<<<16384, 256, 0, stream>>>(tbuf, ln1_w, ln1_b, abuf);

    st_gemm<<<dim3(128, 4), 256, 0, stream>>>(abuf, wqT, nullptr, nullptr, qbuf, nullptr, 512, 512, 0);
    st_gemm<<<dim3(128, 4), 256, 0, stream>>>(abuf, wkT, nullptr, nullptr, kbuf, nullptr, 512, 512, 0);
    st_gemm<<<dim3(128, 4), 256, 0, stream>>>(abuf, wvT, nullptr, nullptr, vbuf, nullptr, 512, 512, 0);

    st_attn<<<dim3(8, 8, 16), 256, 0, stream>>>(qbuf, kbuf, vbuf, abuf);

    st_gemm<<<dim3(128, 4), 256, 0, stream>>>(abuf, woT, bo, tbuf, nullptr, nullptr, 512, 512, 2);

    st_layernorm<<<16384, 256, 0, stream>>>(tbuf, ln3_w, ln3_b, abuf);

    st_gemm_gated<<<dim3(128, 32), 256, 0, stream>>>(abuf, wgT, bg, mbuf);

    st_gemm<<<dim3(128, 4), 256, 0, stream>>>(mbuf, wdT, bd, tbuf, vbuf, nullptr, 2048, 512, 3);

    st_gemm<<<dim3(128, 4), 256, 0, stream>>>(vbuf, poutW, pout_b, (float*)d_out, nullptr, x, 512, 512, 4);
}
